// GCN_16063177687062
// MI455X (gfx1250) — matrix operand order repaired and checked by reading, not compiled, not run on hardware
//
#include <hip/hip_runtime.h>
#include <hip/hip_bf16.h>

#define N_NODES 50000
#define N_EDGES 800000
#define F_IN    1433
#define HID     256
#define N_CLS   7

typedef __attribute__((ext_vector_type(16))) __bf16 v16bf;
typedef __attribute__((ext_vector_type(8)))  float  v8f;

// ---------------------------------------------------------------------------
// init: deg=1.0 (self loop), zero layer-1 aggregation buffer and output
// ---------------------------------------------------------------------------
__global__ void gcn_init_kernel(float* __restrict__ deg,
                                float* __restrict__ agg1,
                                float* __restrict__ out) {
    size_t i = (size_t)blockIdx.x * blockDim.x + threadIdx.x;
    size_t stride = (size_t)gridDim.x * blockDim.x;
    for (size_t j = i; j < (size_t)N_NODES * HID; j += stride) agg1[j] = 0.0f;
    for (size_t j = i; j < (size_t)N_NODES; j += stride) deg[j] = 1.0f;
    for (size_t j = i; j < (size_t)N_NODES * N_CLS; j += stride) out[j] = 0.0f;
}

// ---------------------------------------------------------------------------
// one-time weight conversion to bf16 (B operands pre-laid-out row-major)
// ---------------------------------------------------------------------------
__global__ void gcn_convw1_kernel(const float* __restrict__ W1, __bf16* __restrict__ W1bf) {
    int i = blockIdx.x * blockDim.x + threadIdx.x;
    if (i < F_IN * HID) W1bf[i] = (__bf16)W1[i];
}

// W2 [256,7] -> padded bf16 [256,16] (cols 7..15 = 0)
__global__ void gcn_convw2_kernel(const float* __restrict__ W2, __bf16* __restrict__ W2bf) {
    int i = blockIdx.x * blockDim.x + threadIdx.x;
    if (i < HID * 16) {
        int k = i >> 4, n = i & 15;
        W2bf[i] = (n < N_CLS) ? (__bf16)W2[k * N_CLS + n] : (__bf16)0.0f;
    }
}

// ---------------------------------------------------------------------------
// degree count at target nodes, then dinv = rsqrt(deg) in place
// ---------------------------------------------------------------------------
__global__ void gcn_degree_kernel(const int* __restrict__ col, float* __restrict__ deg) {
    int e = blockIdx.x * blockDim.x + threadIdx.x;
    if (e < N_EDGES) atomicAdd(&deg[col[e]], 1.0f);
}

__global__ void gcn_rsqrt_kernel(float* __restrict__ deg) {
    int i = blockIdx.x * blockDim.x + threadIdx.x;
    if (i < N_NODES) deg[i] = rsqrtf(deg[i]);
}

// ---------------------------------------------------------------------------
// GEMM1: H1[N,256] = X[N,1433] @ W1bf[1433,256], bf16 WMMA, f32 accumulate.
// Block = 256 threads = 8 waves; block owns one 16-row M tile; wave w owns
// two N tiles [32w,32w+16) and [32w+16,32w+32), sharing one A fragment.
// Main K loop (44 iters) is branchless; single peeled tail handles K=1408..1432.
//
// WMMA operand layouts (CDNA5 ISA 7.12.2):
//   A 16x32 bf16 : lane L -> M=L&15 ; VGPR v packs K = (v<4?0:16)+8*(L>>4)+2*(v&3), +1
//   B 32x16 bf16 : lane L -> N=L&15 ; VGPR v packs K = (v<4?0:16)+8*(L>>4)+2*(v&3), +1
//   C/D 16x16 f32: VGPR r -> M=r (lanes 0-15) / M=r+8 (lanes 16-31), N=lane&15
// ---------------------------------------------------------------------------
__global__ __launch_bounds__(256) void gcn_gemm1_kernel(const float* __restrict__ X,
                                                        const __bf16* __restrict__ Wbf,
                                                        float* __restrict__ H) {
    const int wave  = threadIdx.x >> 5;
    const int lane  = threadIdx.x & 31;
    const int half  = lane >> 4;
    const int m0    = blockIdx.x * 16;
    const int n0    = wave * 32;
    const int mrow  = m0 + (lane & 15);
    const int abase = 8 * half;
    const float* __restrict__ xrow = X + (size_t)mrow * F_IN;

    constexpr int KMAIN = (F_IN / 32) * 32;  // 1408

    v8f acc0 = {}, acc1 = {};
    for (int k0 = 0; k0 < KMAIN; k0 += 32) {
        v16bf a;
#pragma unroll
        for (int v = 0; v < 8; ++v) {
            int kk = k0 + ((v < 4) ? 0 : 16) + abase + 2 * (v & 3);
            a[2 * v]     = (__bf16)xrow[kk];
            a[2 * v + 1] = (__bf16)xrow[kk + 1];
        }
        v16bf b0, b1;
#pragma unroll
        for (int s = 0; s < 16; ++s) {
            const int kb = k0 + ((s < 8) ? 0 : 16) + abase + (s & 7);
            const __bf16* __restrict__ brow = Wbf + (size_t)kb * HID + n0 + (lane & 15);
            b0[s] = brow[0];
            b1[s] = brow[16];
        }
        acc0 = __builtin_amdgcn_wmma_f32_16x16x32_bf16(false, a, false, b0,
                                                       (short)0, acc0, false, false);
        acc1 = __builtin_amdgcn_wmma_f32_16x16x32_bf16(false, a, false, b1,
                                                       (short)0, acc1, false, false);
    }

    // peeled branchless tail: K = 1408..1432 valid, rest zero
    {
        v16bf a;
#pragma unroll
        for (int v = 0; v < 8; ++v) {
            int kk  = KMAIN + ((v < 4) ? 0 : 16) + abase + 2 * (v & 3);
            int kc0 = (kk     < F_IN) ? kk     : 0;   // clamp address, mask value
            int kc1 = (kk + 1 < F_IN) ? kk + 1 : 0;
            float f0 = (kk     < F_IN) ? xrow[kc0] : 0.0f;
            float f1 = (kk + 1 < F_IN) ? xrow[kc1] : 0.0f;
            a[2 * v]     = (__bf16)f0;
            a[2 * v + 1] = (__bf16)f1;
        }
        v16bf b0, b1;
#pragma unroll
        for (int s = 0; s < 16; ++s) {
            const int  kb  = KMAIN + ((s < 8) ? 0 : 16) + abase + (s & 7);
            const bool kin = (kb < F_IN);
            const __bf16* __restrict__ brow = Wbf + (size_t)(kin ? kb : 0) * HID + n0 + (lane & 15);
            b0[s] = kin ? brow[0]  : (__bf16)0.0f;
            b1[s] = kin ? brow[16] : (__bf16)0.0f;
        }
        acc0 = __builtin_amdgcn_wmma_f32_16x16x32_bf16(false, a, false, b0,
                                                       (short)0, acc0, false, false);
        acc1 = __builtin_amdgcn_wmma_f32_16x16x32_bf16(false, a, false, b1,
                                                       (short)0, acc1, false, false);
    }

    const int ncol = lane & 15;
#pragma unroll
    for (int r = 0; r < 8; ++r) {
        int mr = m0 + r + 8 * half;
        H[(size_t)mr * HID + n0 + ncol]      = acc0[r];
        H[(size_t)mr * HID + n0 + 16 + ncol] = acc1[r];
    }
}

// ---------------------------------------------------------------------------
// Edge aggregation layer 1: agg[col] += H[row] * dinv[row]*dinv[col]
// One block (256 threads) per edge; one feature per thread.
// ---------------------------------------------------------------------------
__global__ __launch_bounds__(HID) void gcn_scatter1_kernel(const int* __restrict__ row,
                                                           const int* __restrict__ col,
                                                           const float* __restrict__ dinv,
                                                           const float* __restrict__ H,
                                                           float* __restrict__ agg) {
    const int e = blockIdx.x;
    const int r = row[e];
    const int c = col[e];
    const float w = dinv[r] * dinv[c];
    const int f = threadIdx.x;
    atomicAdd(&agg[(size_t)c * HID + f], H[(size_t)r * HID + f] * w);
}

// agg = relu(agg + H*dinv^2 + b1)  (in place)
__global__ void gcn_finalize1_kernel(const float* __restrict__ H,
                                     const float* __restrict__ dinv,
                                     const float* __restrict__ b1,
                                     float* __restrict__ agg) {
    size_t idx = (size_t)blockIdx.x * blockDim.x + threadIdx.x;
    if (idx >= (size_t)N_NODES * HID) return;
    int i = (int)(idx >> 8);        // / HID
    int f = (int)(idx & (HID - 1)); // % HID
    float d = dinv[i];
    float v = agg[idx] + H[idx] * d * d + b1[f];
    agg[idx] = v > 0.0f ? v : 0.0f;
}

// ---------------------------------------------------------------------------
// GEMM2: H2[N,8(pad)] = H1relu[N,256] @ W2bf[256,16(pad)], bf16 WMMA.
// Block = 256 threads = 8 waves; each wave owns one 16-row M tile.
// K=256 (no tail), B pre-padded -> fully branchless.
// ---------------------------------------------------------------------------
__global__ __launch_bounds__(256) void gcn_gemm2_kernel(const float* __restrict__ H,
                                                        const __bf16* __restrict__ Wbf,
                                                        float* __restrict__ O) {
    const int wave  = threadIdx.x >> 5;
    const int lane  = threadIdx.x & 31;
    const int mtile = blockIdx.x * 8 + wave;
    if (mtile * 16 >= N_NODES) return;   // wave-uniform exit: EXEC all-1s at WMMA
    const int half  = lane >> 4;
    const int m0    = mtile * 16;
    const int mrow  = m0 + (lane & 15);
    const int abase = 8 * half;
    const float* __restrict__ hrow = H + (size_t)mrow * HID;

    v8f acc = {};
#pragma unroll
    for (int k0 = 0; k0 < HID; k0 += 32) {
        v16bf a;
#pragma unroll
        for (int v = 0; v < 8; ++v) {
            int kk = k0 + ((v < 4) ? 0 : 16) + abase + 2 * (v & 3);
            a[2 * v]     = (__bf16)hrow[kk];
            a[2 * v + 1] = (__bf16)hrow[kk + 1];
        }
        v16bf b;
#pragma unroll
        for (int s = 0; s < 16; ++s) {
            const int kb = k0 + ((s < 8) ? 0 : 16) + abase + (s & 7);
            b[s] = Wbf[(size_t)kb * 16 + (lane & 15)];
        }
        acc = __builtin_amdgcn_wmma_f32_16x16x32_bf16(false, a, false, b,
                                                      (short)0, acc, false, false);
    }

    const int ncol = lane & 15;
    if (ncol < 8) {
#pragma unroll
        for (int r = 0; r < 8; ++r) {
            int mr = m0 + r + 8 * half;
            O[(size_t)mr * 8 + ncol] = acc[r];
        }
    }
}

// Edge aggregation layer 2: out[col,0:7] += H2[row,0:7] * dinv[row]*dinv[col]
__global__ void gcn_scatter2_kernel(const int* __restrict__ row,
                                    const int* __restrict__ col,
                                    const float* __restrict__ dinv,
                                    const float* __restrict__ H2,
                                    float* __restrict__ out) {
    int e = blockIdx.x * blockDim.x + threadIdx.x;
    if (e >= N_EDGES) return;
    int r = row[e];
    int c = col[e];
    float w = dinv[r] * dinv[c];
#pragma unroll
    for (int f = 0; f < N_CLS; ++f)
        atomicAdd(&out[(size_t)c * N_CLS + f], H2[(size_t)r * 8 + f] * w);
}

// out += H2*dinv^2 + b2
__global__ void gcn_finalize2_kernel(const float* __restrict__ H2,
                                     const float* __restrict__ dinv,
                                     const float* __restrict__ b2,
                                     float* __restrict__ out) {
    int idx = blockIdx.x * blockDim.x + threadIdx.x;
    if (idx >= N_NODES * N_CLS) return;
    int i = idx / N_CLS;
    int f = idx % N_CLS;
    float d = dinv[i];
    out[idx] += H2[(size_t)i * 8 + f] * d * d + b2[f];
}

// ---------------------------------------------------------------------------
extern "C" void kernel_launch(void* const* d_in, const int* in_sizes, int n_in,
                              void* d_out, int out_size, void* d_ws, size_t ws_size,
                              hipStream_t stream) {
    const float* x  = (const float*)d_in[0];
    const int*   ei = (const int*)d_in[1];   // [2, E] flat: row then col
    const float* W1 = (const float*)d_in[2];
    const float* b1 = (const float*)d_in[3];
    const float* W2 = (const float*)d_in[4];
    const float* b2 = (const float*)d_in[5];
    const int* row = ei;
    const int* col = ei + N_EDGES;

    // workspace layout (256B aligned partitions)
    char* ws = (char*)d_ws;
    const size_t OFF_DINV = 0;
    const size_t OFF_H1   = 200192;                                  // align(N*4)
    const size_t OFF_AGG1 = OFF_H1 + (size_t)N_NODES * HID * 4;      //  51,400,192
    const size_t OFF_H2   = OFF_AGG1 + (size_t)N_NODES * HID * 4;    // 102,600,192
    const size_t OFF_W1BF = OFF_H2 + (size_t)N_NODES * 8 * 4;        // 104,200,192
    const size_t OFF_W2BF = OFF_W1BF + (size_t)F_IN * HID * 2;       // 104,933,888 -> pad
    float*  dinv = (float*)(ws + OFF_DINV);
    float*  H1   = (float*)(ws + OFF_H1);
    float*  agg1 = (float*)(ws + OFF_AGG1);
    float*  H2   = (float*)(ws + OFF_H2);
    __bf16* W1bf = (__bf16*)(ws + OFF_W1BF);
    __bf16* W2bf = (__bf16*)(ws + ((OFF_W2BF + 255) & ~(size_t)255));
    float*  out  = (float*)d_out;

    gcn_init_kernel<<<2048, 256, 0, stream>>>(dinv, agg1, out);
    gcn_convw1_kernel<<<(F_IN * HID + 255) / 256, 256, 0, stream>>>(W1, W1bf);
    gcn_convw2_kernel<<<(HID * 16 + 255) / 256, 256, 0, stream>>>(W2, W2bf);
    gcn_degree_kernel<<<(N_EDGES + 255) / 256, 256, 0, stream>>>(col, dinv);
    gcn_rsqrt_kernel<<<(N_NODES + 255) / 256, 256, 0, stream>>>(dinv);

    gcn_gemm1_kernel<<<N_NODES / 16, 256, 0, stream>>>(x, W1bf, H1);
    gcn_scatter1_kernel<<<N_EDGES, HID, 0, stream>>>(row, col, dinv, H1, agg1);
    gcn_finalize1_kernel<<<(int)(((size_t)N_NODES * HID + 255) / 256), 256, 0, stream>>>(
        H1, dinv, b1, agg1);

    gcn_gemm2_kernel<<<(N_NODES / 16 + 7) / 8, 256, 0, stream>>>(agg1, W2bf, H2);
    gcn_scatter2_kernel<<<(N_EDGES + 255) / 256, 256, 0, stream>>>(row, col, dinv, H2, out);
    gcn_finalize2_kernel<<<(N_NODES * N_CLS + 255) / 256, 256, 0, stream>>>(
        H2, dinv, b2, out);
}
